// Refiner_37632503447783
// MI455X (gfx1250) — matrix operand order repaired and checked by reading, not compiled, not run on hardware
//
#include <hip/hip_runtime.h>
#include <hip/hip_bf16.h>
#include <math.h>

typedef __attribute__((ext_vector_type(16))) _Float16 v16h;
typedef __attribute__((ext_vector_type(8)))  _Float16 v8h;
typedef __attribute__((ext_vector_type(8)))  float    v8f;

static constexpr int BB    = 4;
static constexpr int Hf    = 2048;
static constexpr int Wf    = 2048;
static constexpr int HC    = 512;      // quarter res
static constexpr int KSEL  = 5000;     // 80000/16
static constexpr int NERR  = HC * HC;  // 262144
static constexpr int EQCAP = 1024;

// ---------------------------------------------------------------------------
// helpers
// ---------------------------------------------------------------------------
__device__ __forceinline__ float bilerp512(const float* p, float cy, float cx) {
  int iy0 = (int)floorf(cy); float fy = cy - (float)iy0;
  int ix0 = (int)floorf(cx); float fx = cx - (float)ix0;
  int y0 = min(max(iy0, 0), 511), y1 = min(max(iy0 + 1, 0), 511);
  int x0 = min(max(ix0, 0), 511), x1 = min(max(ix0 + 1, 0), 511);
  float a = p[y0 * 512 + x0], b = p[y0 * 512 + x1];
  float c = p[y1 * 512 + x0], d = p[y1 * 512 + x1];
  return (1.f - fy) * ((1.f - fx) * a + fx * b) + fy * ((1.f - fx) * c + fx * d);
}

__device__ __forceinline__ void wave_lds_fence() {
  // wave32 lockstep: waiting on this wave's DS ops makes cross-lane LDS
  // producer->consumer safe without a workgroup barrier.
  asm volatile("s_wait_dscnt 0" ::: "memory");
}

// ---------------------------------------------------------------------------
// WMMA k-step chain, kt is a compile-time constant so all im2col offsets fold.
// Layouts per CDNA5 ISA 7.12.2 (wave32):
//  A 16x32 f16: lane&15 = M row; lane[4]=khi; k = 8*khi + 16*(v>=4) + 2*(v&3)+h
//  B 32x16 f16: lane&15 = N col; k = 8*khi + 16*(v>=4) + 2*(v&3)+h (same as A)
//  C 16x16 f32: lane&15 = N col; vgpr r -> m = r + 8*khi
// Out-of-range M rows / N cols are clamped (garbage, discarded at writeback).
// k >= K9 pads read clamped A garbage times exact-zero padded weights -> 0.
// ---------------------------------------------------------------------------
template <int CIN, int INW, int KT, int KTN>
__device__ __forceinline__ void kSteps(v8f& acc,
                                       const _Float16* __restrict__ in,
                                       const _Float16* __restrict__ w,
                                       int rowBase, int wBase, int khi) {
  if constexpr (KT < KTN) {
    constexpr int K9  = CIN * 9;
    constexpr int K9P = (K9 + 31) & ~31;
    // ---- A fragment: 16 unconditional ds_load_u16, constant offsets ----
    union { v16h v; _Float16 e[16]; } a;
#pragma unroll
    for (int vg = 0; vg < 8; ++vg) {
#pragma unroll
      for (int h = 0; h < 2; ++h) {
        int kc = KT * 32 + ((vg >= 4) ? 16 : 0) + ((vg & 3) << 1) + h; // literal
        int k0 = (kc     < K9) ? kc     : (K9 - 1);                    // literal
        int k1 = (kc + 8 < K9) ? kc + 8 : (K9 - 1);                    // literal
        int o0 = (k0 / 9) * (INW * INW) + ((k0 % 9) / 3) * INW + (k0 % 9) % 3;
        int o1 = (k1 / 9) * (INW * INW) + ((k1 % 9) / 3) * INW + (k1 % 9) % 3;
        a.e[vg * 2 + h] = in[rowBase + (khi ? o1 : o0)];
      }
    }
    // ---- B fragment: two 8-half padded-weight runs at +0 and +16 = 2x ds_load_b128 --
    union { v16h v; v8h q[2]; } bf;
    const _Float16* wp = w + wBase + KT * 32 + khi * 8;    // 16B aligned
    bf.q[0] = *(const v8h*)(wp);
    bf.q[1] = *(const v8h*)(wp + 16);
    acc = __builtin_amdgcn_wmma_f32_16x16x32_f16(false, a.v, false, bf.v,
                                                 (short)0, acc, false, false);
    kSteps<CIN, INW, KT + 1, KTN>(acc, in, w, rowBase, wBase, khi);
  }
}

// conv3x3 VALID + BN + ReLU, f16 in/out via LDS, f32 accumulate in WMMA.
// `w` is LDS weights laid out [COUT][K9P] with K-dim zero padded to K9P.
template <int CIN, int COUT, int INW, int OUTW>
__device__ __forceinline__ void convStage(const _Float16* __restrict__ in,
                                          const _Float16* __restrict__ w,
                                          const float* scale, const float* bias,
                                          _Float16* __restrict__ out, int lane) {
  constexpr int M   = OUTW * OUTW;
  constexpr int K9  = CIN * 9;
  constexpr int K9P = (K9 + 31) & ~31;
  constexpr int MT  = (M + 15) / 16, NT = (COUT + 15) / 16, KT = (K9 + 31) / 32;
  const int lr  = lane & 15;
  const int khi = (lane >> 4) & 1;
#pragma unroll
  for (int mt = 0; mt < MT; ++mt) {
    const int m  = mt * 16 + lr;
    const int mc = (m < M) ? m : (M - 1);           // clamp: discarded rows
    const int oy = mc / OUTW;
    const int rowBase = oy * INW + (mc - oy * OUTW);
#pragma unroll
    for (int nt = 0; nt < NT; ++nt) {
      const int n  = nt * 16 + lr;
      const int nc = (n < COUT) ? n : (COUT - 1);   // clamp: discarded cols
      const int wBase = nc * K9P;
      union { v8f v; float e[8]; } c;
#pragma unroll
      for (int r = 0; r < 8; ++r) c.e[r] = 0.f;
      kSteps<CIN, INW, 0, KT>(c.v, in, w, rowBase, wBase, khi);
      if (n < COUT) {
        float sc = scale[n], bi = bias[n];
#pragma unroll
        for (int r = 0; r < 8; ++r) {
          int mm = mt * 16 + r + (khi ? 8 : 0);
          if (mm < M) {
            float y = c.e[r] * sc + bi;
            y = y > 0.f ? y : 0.f;
            out[n * M + mm] = (_Float16)y;
          }
        }
      }
    }
  }
  wave_lds_fence();
}

// ---------------------------------------------------------------------------
// Top-K: 3-pass radix select on float bits (all err >= 0 -> bits monotonic)
// state per batch (8 u32): [0]=prefix, [1]=kneed/remaining, [2]=thrKey,
//                          [3]=cntGt, [4]=eqCnt
// ---------------------------------------------------------------------------
__global__ void histKernel(const float* __restrict__ err, unsigned* __restrict__ hist,
                           const unsigned* __restrict__ state, int level) {
  __shared__ unsigned h[2048];
  int b = blockIdx.x;
  for (int i = threadIdx.x; i < 2048; i += blockDim.x) h[i] = 0;
  __syncthreads();
  unsigned prefix = state[b * 8 + 0];
  const float* e = err + (size_t)b * NERR;
  for (int i = threadIdx.x; i < NERR; i += blockDim.x) {
    unsigned key = __float_as_uint(e[i]);
    bool ok; unsigned bin;
    if (level == 0)      { ok = true;                        bin = key >> 21; }
    else if (level == 1) { ok = ((key >> 21) == prefix);     bin = (key >> 10) & 2047u; }
    else                 { ok = ((key >> 10) == prefix);     bin = key & 1023u; }
    if (ok) atomicAdd(&h[bin], 1u);
  }
  __syncthreads();
  for (int i = threadIdx.x; i < 2048; i += blockDim.x) hist[b * 2048 + i] = h[i];
}

__global__ void scanKernel(const unsigned* __restrict__ hist, unsigned* __restrict__ state,
                           int level) {
  int b = blockIdx.x;
  if (threadIdx.x != 0) return;
  unsigned kneed = (level == 0) ? (unsigned)KSEL : state[b * 8 + 1];
  int nbins = (level == 2) ? 1024 : 2048;
  unsigned acc = 0; int sel = 0;
  for (int i = nbins - 1; i >= 0; --i) {
    unsigned c = hist[b * 2048 + i];
    if (acc + c >= kneed) { sel = i; break; }
    acc += c;
  }
  state[b * 8 + 1] = kneed - acc;  // still needed within bin `sel`
  if (level == 0)      state[b * 8 + 0] = (unsigned)sel;
  else if (level == 1) state[b * 8 + 0] = (state[b * 8 + 0] << 11) | (unsigned)sel;
  else {
    state[b * 8 + 2] = (state[b * 8 + 0] << 10) | (unsigned)sel;  // exact thr key
    state[b * 8 + 3] = 0;  // cntGt
    state[b * 8 + 4] = 0;  // eqCnt
  }
}

__global__ void compactKernel(const float* __restrict__ err, unsigned* __restrict__ state,
                              int* __restrict__ selIdx, int* __restrict__ eqList) {
  int b = blockIdx.x;
  unsigned thr = state[b * 8 + 2];
  const float* e = err + (size_t)b * NERR;
  for (int i = threadIdx.x; i < NERR; i += blockDim.x) {
    unsigned key = __float_as_uint(e[i]);
    if (key > thr) {
      unsigned pos = atomicAdd(&state[b * 8 + 3], 1u);
      if (pos < (unsigned)KSEL) selIdx[b * KSEL + pos] = i;
    } else if (key == thr) {
      unsigned q = atomicAdd(&state[b * 8 + 4], 1u);
      if (q < (unsigned)EQCAP) eqList[b * EQCAP + q] = i;
    }
  }
}

__global__ void finalizeKernel(const unsigned* __restrict__ state,
                               int* __restrict__ selIdx, int* __restrict__ eqList) {
  int b = blockIdx.x;
  if (threadIdx.x != 0) return;
  int cnt = min((int)state[b * 8 + 4], EQCAP);
  int* eq = eqList + b * EQCAP;
  // insertion sort ascending -> deterministic tie-break by lowest index
  for (int i = 1; i < cnt; ++i) {
    int key = eq[i]; int j = i - 1;
    while (j >= 0 && eq[j] > key) { eq[j + 1] = eq[j]; --j; }
    eq[j + 1] = key;
  }
  int rem = (int)state[b * 8 + 1];
  if (rem > KSEL) rem = KSEL;
  int base = KSEL - rem;
  for (int j = 0; j < rem; ++j) {
    int v = (cnt > 0) ? eq[min(j, cnt - 1)] : 0;
    selIdx[b * KSEL + base + j] = v;
  }
}

__global__ void zeroRefKernel(float* __restrict__ ref) {
  size_t t = (size_t)blockIdx.x * blockDim.x + threadIdx.x;
  if (t < (size_t)BB * NERR) ref[t] = 0.f;
}

__global__ void scatterRefKernel(const int* __restrict__ selIdx, float* __restrict__ ref) {
  int t = blockIdx.x * blockDim.x + threadIdx.x;
  if (t < BB * KSEL) {
    int b = t / KSEL;
    ref[(size_t)b * NERR + selIdx[t]] = 1.0f;
  }
}

// ---------------------------------------------------------------------------
// Bilinear x4 upsample of [pha, fgr] -> out base image (268 MB streaming)
// ---------------------------------------------------------------------------
__global__ void upsample4Kernel(const float* __restrict__ pha, const float* __restrict__ fgr,
                                float* __restrict__ outPha, float* __restrict__ outFgr) {
  size_t t = (size_t)blockIdx.x * blockDim.x + threadIdx.x;
  if (t >= (size_t)BB * 4 * Hf * Wf) return;
  int ox = (int)(t & 2047);
  size_t r = t >> 11;
  int oy = (int)(r & 2047); r >>= 11;
  int c  = (int)(r & 3);
  int b  = (int)(r >> 2);
  float cy = ((float)oy + 0.5f) * 0.25f - 0.5f;
  float cx = ((float)ox + 0.5f) * 0.25f - 0.5f;
  const float* p = (c == 0) ? pha + (size_t)b * NERR
                            : fgr + ((size_t)(b * 3 + c - 1)) * NERR;
  float v = bilerp512(p, cy, cx);
  if (c == 0) outPha[((size_t)b * Hf + oy) * Wf + ox] = v;
  else        outFgr[(((size_t)b * 3 + (c - 1)) * Hf + oy) * Wf + ox] = v;
}

// ---------------------------------------------------------------------------
// Patch refinement: one patch per wave, 4 WMMA conv stages
// ---------------------------------------------------------------------------
__global__ __launch_bounds__(128) void refineKernel(
    const float* __restrict__ src, const float* __restrict__ bgr,
    const float* __restrict__ pha, const float* __restrict__ fgr,
    const float* __restrict__ hid,
    const float* __restrict__ w1, const float* __restrict__ w2,
    const float* __restrict__ w3, const float* __restrict__ w4,
    const float* __restrict__ b4,
    const float* __restrict__ g1, const float* __restrict__ be1,
    const float* __restrict__ m1, const float* __restrict__ vv1,
    const float* __restrict__ g2, const float* __restrict__ be2,
    const float* __restrict__ m2, const float* __restrict__ vv2,
    const float* __restrict__ g3, const float* __restrict__ be3,
    const float* __restrict__ m3, const float* __restrict__ vv3,
    const int* __restrict__ selIdx,
    float* __restrict__ outPha, float* __restrict__ outFgr) {
  // weights zero-padded in K to multiples of 32 halves (exact-zero k-pad)
  __shared__ __attribute__((aligned(16))) _Float16 sW1[24 * 384];
  __shared__ __attribute__((aligned(16))) _Float16 sW2[16 * 224];
  __shared__ __attribute__((aligned(16))) _Float16 sW3[12 * 224];
  __shared__ __attribute__((aligned(16))) _Float16 sW4[4 * 128];
  __shared__ float    sBN[108];     // s1[24] b1[24] s2[16] b2[16] s3[12] b3[12] b4[4]
  __shared__ __attribute__((aligned(16))) _Float16 sA[4][2688];  // per-wave inputs
  __shared__ __attribute__((aligned(16))) _Float16 sB[4][864];   // per-wave outputs

  const int tid = threadIdx.x;
  // --- block init: weights -> padded f16 LDS, BN folded scale/bias ---
  for (int i = tid; i < 24 * 384; i += 128) {
    int n = i / 384, k = i - n * 384;
    sW1[i] = (_Float16)((k < 378) ? w1[n * 378 + k] : 0.f);
  }
  for (int i = tid; i < 16 * 224; i += 128) {
    int n = i / 224, k = i - n * 224;
    sW2[i] = (_Float16)((k < 216) ? w2[n * 216 + k] : 0.f);
  }
  for (int i = tid; i < 12 * 224; i += 128) {
    int n = i / 224, k = i - n * 224;
    sW3[i] = (_Float16)((k < 198) ? w3[n * 198 + k] : 0.f);
  }
  for (int i = tid; i < 4 * 128; i += 128) {
    int n = i / 128, k = i - n * 128;
    sW4[i] = (_Float16)((k < 108) ? w4[n * 108 + k] : 0.f);
  }
  if (tid < 24)      { float s = g1[tid] * rsqrtf(vv1[tid] + 1e-5f);
                       sBN[tid] = s; sBN[24 + tid] = be1[tid] - m1[tid] * s; }
  else if (tid < 40) { int i = tid - 24; float s = g2[i] * rsqrtf(vv2[i] + 1e-5f);
                       sBN[48 + i] = s; sBN[64 + i] = be2[i] - m2[i] * s; }
  else if (tid < 52) { int i = tid - 40; float s = g3[i] * rsqrtf(vv3[i] + 1e-5f);
                       sBN[80 + i] = s; sBN[92 + i] = be3[i] - m3[i] * s; }
  else if (tid < 56) { sBN[104 + tid - 52] = b4[tid - 52]; }
  __syncthreads();

  const int wv = tid >> 5, lane = tid & 31;
  _Float16* A  = sA[wv];
  _Float16* Bb = sB[wv];
  const int pTotal = BB * KSEL;

  for (int p = blockIdx.x * 4 + wv; p < pTotal; p += gridDim.x * 4) {
    int b  = p / KSEL;
    int fi = selIdx[p];
    int ih = fi >> 9;        // / 512
    int iw = fi & 511;

    // --- stage 0: gather conv1 input [42][8][8] into A ---
    for (int e = lane; e < 42 * 64; e += 32) {
      int c = e >> 6, pos = e & 63, j = pos >> 3, i = pos & 7;
      int y = ih * 2 + j - 3, x = iw * 2 + i - 3;
      float val = 0.f;
      if ((unsigned)y < 1024u && (unsigned)x < 1024u) {
        if (c < 36) {
          const float* ptr = (c < 32)  ? hid + ((size_t)(b * 32 + c)) * NERR
                           : (c == 32) ? pha + (size_t)b * NERR
                                       : fgr + ((size_t)(b * 3 + c - 33)) * NERR;
          val = bilerp512(ptr, (float)y * 0.5f - 0.25f, (float)x * 0.5f - 0.25f);
        } else {
          int cc = c - 36;
          const float* ptr = (cc < 3) ? src + ((size_t)(b * 3 + cc)) * (size_t)(Hf * Wf)
                                      : bgr + ((size_t)(b * 3 + cc - 3)) * (size_t)(Hf * Wf);
          const float* r0 = ptr + (size_t)(2 * y) * Wf + 2 * x;
          val = 0.25f * (r0[0] + r0[1] + r0[Wf] + r0[Wf + 1]);
        }
      }
      A[e] = (_Float16)val;
    }
    wave_lds_fence();

    // --- conv1: 42ch 8x8 -> 24ch 6x6 (BN+ReLU) ---
    convStage<42, 24, 8, 6>(A, sW1, &sBN[0], &sBN[24], Bb, lane);
    // --- conv2: 24ch 6x6 -> 16ch 4x4 (BN+ReLU), out -> A[0..255] ---
    convStage<24, 16, 6, 4>(Bb, sW2, &sBN[48], &sBN[64], A, lane);

    // --- stage: build conv3 input [22][8][8] at A+256 ---
    for (int e = lane; e < 22 * 64; e += 32) {
      int c = e >> 6, pos = e & 63, j = pos >> 3, i = pos & 7;
      _Float16 val;
      if (c < 16) {
        val = A[c * 16 + (j >> 1) * 4 + (i >> 1)];     // nearest x2 of conv2 out
      } else {
        int cc = c - 16;
        int y = ih * 4 + j - 2, x = iw * 4 + i - 2;
        float f = 0.f;
        if ((unsigned)y < 2048u && (unsigned)x < 2048u) {
          const float* ptr = (cc < 3) ? src + ((size_t)(b * 3 + cc)) * (size_t)(Hf * Wf)
                                      : bgr + ((size_t)(b * 3 + cc - 3)) * (size_t)(Hf * Wf);
          f = ptr[(size_t)y * Wf + x];
        }
        val = (_Float16)f;
      }
      A[256 + e] = val;
    }
    wave_lds_fence();

    // --- conv3: 22ch 8x8 -> 12ch 6x6 (BN+ReLU) ---
    convStage<22, 12, 8, 6>(A + 256, sW3, &sBN[80], &sBN[92], Bb, lane);

    // --- conv4: 12ch 6x6 -> 4ch 4x4 (+b4), scatter to out ---
    {
      const int lr = lane & 15, khi = (lane >> 4) & 1;
      const int mc = lr;                 // M=16, all rows valid
      const int oy4 = mc >> 2;
      const int rowBase = oy4 * 6 + (mc & 3);
      const int nc = (lr < 4) ? lr : 3;  // clamp N (COUT=4)
      union { v8f v; float e[8]; } c;
#pragma unroll
      for (int r = 0; r < 8; ++r) c.e[r] = 0.f;
      kSteps<12, 6, 0, 4>(c.v, Bb, sW4, rowBase, nc * 128, khi);  // K9=108 -> 4 steps
      int n = lr;
      if (n < 4) {
        float bias = sBN[104 + n];
#pragma unroll
        for (int r = 0; r < 8; ++r) {
          int m = r + (khi ? 8 : 0);          // 0..15, always valid (M=16)
          int py = m >> 2, px = m & 3;
          float val = c.e[r] + bias;
          size_t oy = (size_t)ih * 4 + py, ox = (size_t)iw * 4 + px;
          if (n == 0) outPha[((size_t)b * Hf + oy) * Wf + ox] = val;
          else        outFgr[(((size_t)b * 3 + (n - 1)) * Hf + oy) * Wf + ox] = val;
        }
      }
    }
  }
}

// ---------------------------------------------------------------------------
// launch
// ---------------------------------------------------------------------------
extern "C" void kernel_launch(void* const* d_in, const int* in_sizes, int n_in,
                              void* d_out, int out_size, void* d_ws, size_t ws_size,
                              hipStream_t stream) {
  const float* src = (const float*)d_in[0];
  const float* bgr = (const float*)d_in[1];
  const float* pha = (const float*)d_in[2];
  const float* fgr = (const float*)d_in[3];
  const float* err = (const float*)d_in[4];
  const float* hid = (const float*)d_in[5];
  const float* w1  = (const float*)d_in[6];
  const float* w2  = (const float*)d_in[7];
  const float* w3  = (const float*)d_in[8];
  const float* w4  = (const float*)d_in[9];
  const float* b4  = (const float*)d_in[10];
  const float* g1  = (const float*)d_in[11];
  const float* be1 = (const float*)d_in[12];
  const float* m1  = (const float*)d_in[13];
  const float* v1  = (const float*)d_in[14];
  const float* g2  = (const float*)d_in[15];
  const float* be2 = (const float*)d_in[16];
  const float* m2  = (const float*)d_in[17];
  const float* v2  = (const float*)d_in[18];
  const float* g3  = (const float*)d_in[19];
  const float* be3 = (const float*)d_in[20];
  const float* m3  = (const float*)d_in[21];
  const float* v3  = (const float*)d_in[22];

  float* outPha = (float*)d_out;                                   // B*1*H*W
  float* outFgr = outPha + (size_t)BB * Hf * Wf;                   // B*3*H*W
  float* outRef = outFgr + (size_t)BB * 3 * Hf * Wf;               // B*512*512

  // workspace layout
  int*      selIdx = (int*)d_ws;                      // B*K
  unsigned* hist   = (unsigned*)(selIdx + BB * KSEL); // B*2048
  unsigned* state  = hist + BB * 2048;                // B*8
  int*      eqList = (int*)(state + BB * 8);          // B*EQCAP

  // --- top-K radix select (3 passes over float bits) ---
  histKernel<<<BB, 1024, 0, stream>>>(err, hist, state, 0);
  scanKernel<<<BB, 64, 0, stream>>>(hist, state, 0);
  histKernel<<<BB, 1024, 0, stream>>>(err, hist, state, 1);
  scanKernel<<<BB, 64, 0, stream>>>(hist, state, 1);
  histKernel<<<BB, 1024, 0, stream>>>(err, hist, state, 2);
  scanKernel<<<BB, 64, 0, stream>>>(hist, state, 2);
  compactKernel<<<BB, 1024, 0, stream>>>(err, state, selIdx, eqList);
  finalizeKernel<<<BB, 64, 0, stream>>>(state, selIdx, eqList);

  // --- ref one-hot map ---
  zeroRefKernel<<<(BB * NERR + 255) / 256, 256, 0, stream>>>(outRef);
  scatterRefKernel<<<(BB * KSEL + 255) / 256, 256, 0, stream>>>(selIdx, outRef);

  // --- base image: x4 bilinear upsample of [pha,fgr] ---
  upsample4Kernel<<<(int)(((size_t)BB * 4 * Hf * Wf) / 256), 256, 0, stream>>>(
      pha, fgr, outPha, outFgr);

  // --- WMMA patch refinement + scatter ---
  refineKernel<<<512, 128, 0, stream>>>(src, bgr, pha, fgr, hid,
                                        w1, w2, w3, w4, b4,
                                        g1, be1, m1, v1,
                                        g2, be2, m2, v2,
                                        g3, be3, m3, v3,
                                        selIdx, outPha, outFgr);
  (void)in_sizes; (void)n_in; (void)out_size; (void)ws_size;
}
